// LSTM_22351009809226
// MI455X (gfx1250) — matrix operand order repaired and checked by reading, not compiled, not run on hardware
//
#include <hip/hip_runtime.h>
#include <math.h>

// ---------------------------------------------------------------------------
// Problem constants (from reference): B=32, Np1=33, N=32, S=128, F=4, H=512
// ---------------------------------------------------------------------------
#define BB   32
#define NP1  33
#define NN   32
#define SS   128
#define FF   4
#define HH   512
#define MM   (BB * NN)        // 1024 effective LSTM batch
#define G4H  (4 * HH)         // 2048 gate dimension

typedef __attribute__((ext_vector_type(16))) __bf16 v16bf;
typedef __attribute__((ext_vector_type(8)))  float  v8f;

union FragBF {
    v16bf v;
    uint4 q[2];
};

// Per-K-chunk fragment set: two A fragments (32 M-rows) + 4 gate B fragments.
struct Frags {
    FragBF a0, a1;
    FragBF b[4];
};

__device__ __forceinline__ unsigned short f32_to_bf16(float f) {
    union { float f; unsigned int u; } v; v.f = f;
    unsigned int u = v.u;
    u += 0x7FFFu + ((u >> 16) & 1u);   // round-to-nearest-even
    return (unsigned short)(u >> 16);
}

__device__ __forceinline__ float bf16_to_f32(unsigned short h) {
    union { unsigned int u; float f; } v; v.u = ((unsigned int)h) << 16;
    return v.f;
}

__device__ __forceinline__ float sigmoidf(float x) {
    return 1.0f / (1.0f + __expf(-x));
}

// ---------------------------------------------------------------------------
// Kernel 1: convert W_ih / W_hh (f32 row-major [4H][H]) to bf16; fold biases.
// ---------------------------------------------------------------------------
__global__ void convert_weights(const float* __restrict__ Wih,
                                const float* __restrict__ Whh,
                                const float* __restrict__ bih,
                                const float* __restrict__ bhh,
                                unsigned short* __restrict__ WihB,
                                unsigned short* __restrict__ WhhB,
                                float* __restrict__ bias)
{
    const int NW = G4H * HH;                 // 1,048,576 per matrix
    int i = blockIdx.x * blockDim.x + threadIdx.x;
    if (i < NW) {
        WihB[i] = f32_to_bf16(Wih[i]);
    } else if (i < 2 * NW) {
        int j = i - NW;
        WhhB[j] = f32_to_bf16(Whh[j]);
    } else if (i < 2 * NW + G4H) {
        int j = i - 2 * NW;
        bias[j] = bih[j] + bhh[j];
    }
}

// ---------------------------------------------------------------------------
// Kernel 2: zero initial h (bf16, both ping-pong buffers) and c (f32).
// ---------------------------------------------------------------------------
__global__ void init_state(unsigned short* __restrict__ h0,
                           unsigned short* __restrict__ h1,
                           float* __restrict__ c)
{
    int i = blockIdx.x * blockDim.x + threadIdx.x;   // exactly M*H threads
    h0[i] = 0;
    h1[i] = 0;
    c[i]  = 0.0f;
}

// ---------------------------------------------------------------------------
// Kernel 3: h_in[s][m][h] = tanh(xs[m][s][:] . W_lin[h][:] + b_lin[h]) -> bf16
// Each thread produces two consecutive h outputs (one packed u32 store).
// ---------------------------------------------------------------------------
__global__ void prep_hin(const float* __restrict__ x,
                         const float* __restrict__ Wlin,
                         const float* __restrict__ blin,
                         unsigned short* __restrict__ hin)
{
    const int HP = HH / 2;
    size_t tid = (size_t)blockIdx.x * blockDim.x + threadIdx.x;
    int h2  = (int)(tid % HP);
    size_t rem = tid / HP;
    int m   = (int)(rem % MM);
    int s   = (int)(rem / MM);
    int b   = m / NN;
    int nseq = m % NN;

    const float4 xv = *(const float4*)(x + (((size_t)(b * NP1 + nseq) * SS + s) * FF));
    int h0 = h2 * 2;
    const float4 w0 = *(const float4*)(Wlin + (size_t)h0 * FF);
    const float4 w1 = *(const float4*)(Wlin + (size_t)(h0 + 1) * FF);

    float v0 = tanhf(xv.x * w0.x + xv.y * w0.y + xv.z * w0.z + xv.w * w0.w + blin[h0]);
    float v1 = tanhf(xv.x * w1.x + xv.y * w1.y + xv.z * w1.z + xv.w * w1.w + blin[h0 + 1]);

    unsigned int packed = (unsigned int)f32_to_bf16(v0) |
                          ((unsigned int)f32_to_bf16(v1) << 16);
    *(unsigned int*)(hin + tid * 2) = packed;   // hin[(s*M + m)*H + h0]
}

// ---------------------------------------------------------------------------
// Fragment loader for one 32-wide K-chunk (kk = 0..31; kk<16 -> x-part,
// kk>=16 -> h-part). All per-lane loads are contiguous 16B from row-major
// global memory, matching the documented wave32 WMMA A/B VGPR layouts.
// ---------------------------------------------------------------------------
__device__ __forceinline__ void load_frags(Frags& f,
    const unsigned short* __restrict__ Ax,   // hin_t  [M][H]
    const unsigned short* __restrict__ Ah,   // hprev  [M][H]
    const unsigned short* __restrict__ Wi,   // W_ih   [4H][H] bf16
    const unsigned short* __restrict__ Wh,   // W_hh   [4H][H] bf16
    int kk, int arow0, int arow1, int akoff, int ncol, int bkoff)
{
    const unsigned short* A = (kk < 16) ? Ax : Ah;
    const unsigned short* W = (kk < 16) ? Wi : Wh;
    const int k = (kk & 15) * 32;

    // A fragments: lane L holds row M=L%16; K halves selected by L/16.
    const unsigned short* a0p = A + (size_t)arow0 * HH + k + akoff;
    const unsigned short* a1p = A + (size_t)arow1 * HH + k + akoff;
    f.a0.q[0] = *(const uint4*)(a0p);
    f.a0.q[1] = *(const uint4*)(a0p + 16);
    f.a1.q[0] = *(const uint4*)(a1p);
    f.a1.q[1] = *(const uint4*)(a1p + 16);

    // B fragments: lane L holds col N=L%16, K (L/16)*8..+7 and 16+(L/16)*8..+7.
    #pragma unroll
    for (int g = 0; g < 4; ++g) {
        const unsigned short* wptr = W + (size_t)(g * HH + ncol) * HH + k + bkoff;
        f.b[g].q[0] = *(const uint4*)(wptr);
        f.b[g].q[1] = *(const uint4*)(wptr + 16);
    }
}

// ---------------------------------------------------------------------------
// Kernel 4: one LSTM timestep.
//   gates[m][g*H+n] = sum_k hin_t[m][k]*W_ih[g*H+n][k]
//                   + sum_k hprev[m][k]*W_hh[g*H+n][k] + bias[g*H+n]
// Each wave computes a 32(M) x 16(N) tile of H-space for ALL 4 gates
// (8 accumulators) with explicit double-buffered fragment prefetch so loads
// for K-chunk kk+1 are in flight while the 8 WMMAs of chunk kk execute.
// Block mapping: all 8 waves of a block share the same n-tile (weight reuse
// through WGP-local cache), different m-tiles.
// ---------------------------------------------------------------------------
__global__ void __launch_bounds__(256)
lstm_step(const unsigned short* __restrict__ hin_t,   // [M][H] bf16 (t slice)
          const unsigned short* __restrict__ hprev,   // [M][H] bf16
          unsigned short* __restrict__ hnext,         // [M][H] bf16
          float* __restrict__ c,                      // [M][H] f32 (in place)
          const unsigned short* __restrict__ WihB,    // [4H][H] bf16
          const unsigned short* __restrict__ WhhB,    // [4H][H] bf16
          const float* __restrict__ bias)             // [4H] f32
{
    const int lane = threadIdx.x & 31;
    const int wv   = threadIdx.x >> 5;                // wave in block: 0..7
    const int nTile  = blockIdx.x >> 2;               // 0..31  (H/16)
    const int mTile  = ((blockIdx.x & 3) << 3) + wv;  // 0..31  (M/32)
    const int mBase  = mTile * 32;
    const int nBase  = nTile * 16;

    const int arow0 = mBase + (lane & 15);            // first 16 M-rows
    const int arow1 = arow0 + 16;                     // second 16 M-rows
    const int akoff = (lane >> 4) * 8;
    const int ncol  = nBase + (lane & 15);
    const int bkoff = (lane >> 4) * 8;

    v8f acc[2][4] = {};
    Frags buf[2];

    load_frags(buf[0], hin_t, hprev, WihB, WhhB, 0,
               arow0, arow1, akoff, ncol, bkoff);

    #pragma unroll
    for (int kk = 0; kk < 32; ++kk) {
        if (kk + 1 < 32)
            load_frags(buf[(kk + 1) & 1], hin_t, hprev, WihB, WhhB, kk + 1,
                       arow0, arow1, akoff, ncol, bkoff);
        Frags& f = buf[kk & 1];
        #pragma unroll
        for (int g = 0; g < 4; ++g)
            acc[0][g] = __builtin_amdgcn_wmma_f32_16x16x32_bf16(
                false, f.a0.v, false, f.b[g].v, (short)0, acc[0][g], false, false);
        #pragma unroll
        for (int g = 0; g < 4; ++g)
            acc[1][g] = __builtin_amdgcn_wmma_f32_16x16x32_bf16(
                false, f.a1.v, false, f.b[g].v, (short)0, acc[1][g], false, false);
    }

    // Per-column biases (same for all rows a lane owns).
    const float bi = bias[0 * HH + ncol];
    const float bff = bias[1 * HH + ncol];
    const float bg = bias[2 * HH + ncol];
    const float bo = bias[3 * HH + ncol];

    // C/D layout: VGPR r, lane l -> M = base + r + 8*(l>=16), N = nBase + l%16
    #pragma unroll
    for (int mf = 0; mf < 2; ++mf) {
        const int mrow0 = mBase + mf * 16 + ((lane >> 4) << 3);
        #pragma unroll
        for (int r = 0; r < 8; ++r) {
            const int m = mrow0 + r;
            const size_t idx = (size_t)m * HH + ncol;
            const float iv = sigmoidf(acc[mf][0][r] + bi);
            const float fv = sigmoidf(acc[mf][1][r] + bff);
            const float gv = tanhf(acc[mf][2][r] + bg);
            const float ov = sigmoidf(acc[mf][3][r] + bo);
            const float cn = fv * c[idx] + iv * gv;
            c[idx] = cn;
            hnext[idx] = f32_to_bf16(ov * tanhf(cn));
        }
    }
}

// ---------------------------------------------------------------------------
// Kernel 5: out[m] = hT[m][:] . W_pred[0][:] + b_pred   (M=1024 dots of 512)
// ---------------------------------------------------------------------------
__global__ void predict(const unsigned short* __restrict__ hT,
                        const float* __restrict__ Wpred,
                        const float* __restrict__ bpred,
                        float* __restrict__ out)
{
    int m = blockIdx.x * blockDim.x + threadIdx.x;
    if (m >= MM) return;
    const unsigned short* hp = hT + (size_t)m * HH;
    float acc = 0.0f;
    for (int k = 0; k < HH; ++k)
        acc += bf16_to_f32(hp[k]) * Wpred[k];
    out[m] = acc + bpred[0];
}

// ---------------------------------------------------------------------------
// Launch
// ---------------------------------------------------------------------------
extern "C" void kernel_launch(void* const* d_in, const int* in_sizes, int n_in,
                              void* d_out, int out_size, void* d_ws, size_t ws_size,
                              hipStream_t stream)
{
    (void)in_sizes; (void)n_in; (void)out_size; (void)ws_size;

    const float* x      = (const float*)d_in[0];
    /* d_in[1] = mask: all-false, unused by the reference computation */
    const float* W_lin  = (const float*)d_in[2];
    const float* b_lin  = (const float*)d_in[3];
    const float* W_ih   = (const float*)d_in[4];
    const float* W_hh   = (const float*)d_in[5];
    const float* b_ih   = (const float*)d_in[6];
    const float* b_hh   = (const float*)d_in[7];
    const float* W_pred = (const float*)d_in[8];
    const float* b_pred = (const float*)d_in[9];
    float* out = (float*)d_out;

    // ---- workspace carve-up (all offsets 256B-aligned) ----
    char* ws = (char*)d_ws;
    const size_t HIN_B  = (size_t)SS * MM * HH * sizeof(unsigned short); // 128 MiB
    const size_t HBUF_B = (size_t)MM * HH * sizeof(unsigned short);      // 1 MiB
    const size_t CBUF_B = (size_t)MM * HH * sizeof(float);               // 2 MiB
    const size_t WBUF_B = (size_t)G4H * HH * sizeof(unsigned short);     // 2 MiB

    size_t off = 0;
    unsigned short* hin   = (unsigned short*)(ws + off); off += HIN_B;
    unsigned short* hbuf0 = (unsigned short*)(ws + off); off += HBUF_B;
    unsigned short* hbuf1 = (unsigned short*)(ws + off); off += HBUF_B;
    float*          cbuf  = (float*)         (ws + off); off += CBUF_B;
    unsigned short* WihB  = (unsigned short*)(ws + off); off += WBUF_B;
    unsigned short* WhhB  = (unsigned short*)(ws + off); off += WBUF_B;
    float*          bias  = (float*)         (ws + off); off += (size_t)G4H * sizeof(float);
    unsigned short* hbuf[2] = { hbuf0, hbuf1 };

    // 1) weights -> bf16, fold biases
    {
        const int total = 2 * G4H * HH + G4H;
        convert_weights<<<(total + 255) / 256, 256, 0, stream>>>(
            W_ih, W_hh, b_ih, b_hh, WihB, WhhB, bias);
    }

    // 2) zero h0/h1/c   (exactly M*H = 524288 threads)
    init_state<<<(MM * HH) / 256, 256, 0, stream>>>(hbuf0, hbuf1, cbuf);

    // 3) h_in = tanh(xs @ W_lin^T + b_lin) in bf16, layout [S][M][H]
    {
        const size_t total = (size_t)SS * MM * (HH / 2);   // 2 outputs/thread
        prep_hin<<<(unsigned int)(total / 256), 256, 0, stream>>>(
            x, W_lin, b_lin, hin);
    }

    // 4) 128 recurrent steps; ping-pong h, in-place c
    //    grid: 128 blocks x 8 waves = 1024 waves = (M/32)*(H/16) tiles
    for (int t = 0; t < SS; ++t) {
        lstm_step<<<128, 256, 0, stream>>>(
            hin + (size_t)t * MM * HH,
            hbuf[t & 1], hbuf[(t + 1) & 1],
            cbuf, WihB, WhhB, bias);
    }

    // 5) final projection (last write went to hbuf[S & 1] == hbuf0)
    predict<<<(MM + 255) / 256, 256, 0, stream>>>(
        hbuf[SS & 1], W_pred, b_pred, out);
}
